// CARAFE_31164282700244
// MI455X (gfx1250) — matrix operand order repaired and checked by reading, not compiled, not run on hardware
//
#include <hip/hip_runtime.h>
#include <cmath>

typedef _Float16 v16h __attribute__((ext_vector_type(16)));
typedef _Float16 h2   __attribute__((ext_vector_type(2)));
typedef float    v8f  __attribute__((ext_vector_type(8)));
typedef float    f4   __attribute__((ext_vector_type(4)));

#define B_    8
#define C_    128
#define H_    64
#define W_    64
#define MID_  64
#define OC_   100
#define OCP_  128        // OC padded to full m-tiles
#define K3_   576        // 9 taps * 64 ch, reordered k' = tap*64 + c
#define HW_   4096
#define NPIX_ 32768      // B*H*W
#define CSTR_ 72         // padded channel stride of transposed conv3 tile (144 B, 16B-aligned)

// ---------------------------------------------------------------------------
// CDNA5 async global->LDS copy (ASYNCcnt path), per ISA 10.x / 15.18
// ---------------------------------------------------------------------------
__device__ __forceinline__ void async_ld_b128(unsigned int lds_byte_off, const void* gaddr) {
    asm volatile("global_load_async_to_lds_b128 %0, %1, off"
                 :: "v"(lds_byte_off), "v"(gaddr) : "memory");
}
__device__ __forceinline__ void wait_async0() {
    asm volatile("s_wait_asynccnt 0x0" ::: "memory");
}

// ---------------------------------------------------------------------------
// K0: weights -> f16. comp_w kept [64][128]; enc_w reordered to [128][tap*64+c]
// (channel-fastest K so conv3 B-fragment K-runs are contiguous), padded rows 0.
// ---------------------------------------------------------------------------
__global__ void wconv_kernel(const float* __restrict__ comp_w,
                             const float* __restrict__ enc_w,
                             _Float16* __restrict__ w1,
                             _Float16* __restrict__ w2) {
    int i = blockIdx.x * 256 + threadIdx.x;
    if (i < MID_ * C_) w1[i] = (_Float16)comp_w[i];
    if (i < OCP_ * K3_) {
        int m   = i / K3_;
        int kp  = i - m * K3_;
        int tap = kp >> 6;          // 0..8 = ky*3+kx
        int c   = kp & 63;
        w2[i] = (m < OC_) ? (_Float16)enc_w[m * K3_ + c * 9 + tap] : (_Float16)0.f;
    }
}

// ---------------------------------------------------------------------------
// K1: 1x1 conv WMMA GEMM  D[64, 32768] = W[64,128] * X[128, 32768]
// Weights staged via global_load_async_to_lds_b128 (overlaps with x conversion)
// ---------------------------------------------------------------------------
__global__ __launch_bounds__(256) void conv1_kernel(const float* __restrict__ x,
                                                    const _Float16* __restrict__ w1,
                                                    float* __restrict__ y1) {
    __shared__ _Float16 sw[64 * 136];     // [m][k], padded stride (async dest)
    __shared__ _Float16 sx[128 * 136];    // [pix][k]
    const int tid = threadIdx.x;
    const int p0  = blockIdx.x * 128;
    const int b   = p0 >> 12;
    const int hw0 = p0 & 4095;

    // async-stage weights: 64 rows x 16 chunks of 16B, padded LDS row stride
    for (int e = tid; e < 1024; e += 256) {
        int m = e >> 4, ch = e & 15;
        unsigned int loff = (unsigned int)(uintptr_t)&sw[m * 136 + ch * 8];
        async_ld_b128(loff, w1 + m * 128 + ch * 8);
    }
    // x tile: fp32 -> f16 through VALU (overlaps the async weight DMA)
    const size_t xbase = ((size_t)b * C_) * HW_ + hw0;
    for (int e = tid; e < 128 * 128; e += 256) {
        int c = e >> 7, pix = e & 127;
        sx[pix * 136 + c] = (_Float16)x[xbase + (size_t)c * HW_ + pix];
    }
    wait_async0();
    __syncthreads();

    const int wave = tid >> 5, lane = tid & 31;
    const int nloc = wave * 16;
    const int row  = lane & 15;
    const int half = lane >> 4;

    for (int mt = 0; mt < 4; ++mt) {
        v8f acc = {0.f, 0.f, 0.f, 0.f, 0.f, 0.f, 0.f, 0.f};
        for (int ks = 0; ks < 4; ++ks) {
            const int k0 = ks * 32;
            v16h a, bf;
#pragma unroll
            for (int v = 0; v < 8; ++v) {
                int ka = k0 + half * 16 + v * 2;
                h2 pa = *(const h2*)&sw[(mt * 16 + row) * 136 + ka];
                a[2 * v] = pa[0]; a[2 * v + 1] = pa[1];
                int kb = k0 + half * 16 + v * 2;
                h2 pb = *(const h2*)&sx[(nloc + row) * 136 + kb];
                bf[2 * v] = pb[0]; bf[2 * v + 1] = pb[1];
            }
            acc = __builtin_amdgcn_wmma_f32_16x16x32_f16(false, a, false, bf,
                                                         (short)0, acc, false, false);
        }
#pragma unroll
        for (int r = 0; r < 8; ++r) {
            int m = mt * 16 + r + half * 8;
            y1[((size_t)b * MID_ + m) * HW_ + hw0 + nloc + row] = acc[r];
        }
    }
}

// ---------------------------------------------------------------------------
// K2: per-channel batch stats -> fused scale/shift
// ---------------------------------------------------------------------------
__global__ __launch_bounds__(256) void bnstats_kernel(const float* __restrict__ y, int C,
                                                      const float* __restrict__ gamma,
                                                      const float* __restrict__ beta,
                                                      float* __restrict__ scale,
                                                      float* __restrict__ shift) {
    __shared__ float rs[256], rq[256];
    const int c = blockIdx.x, tid = threadIdx.x;
    float s = 0.f, q = 0.f;
    for (int b = 0; b < B_; ++b) {
        const float* p = y + ((size_t)b * C + c) * HW_;
        for (int i = tid; i < HW_; i += 256) { float v = p[i]; s += v; q += v * v; }
    }
    rs[tid] = s; rq[tid] = q;
    __syncthreads();
    for (int off = 128; off > 0; off >>= 1) {
        if (tid < off) { rs[tid] += rs[tid + off]; rq[tid] += rq[tid + off]; }
        __syncthreads();
    }
    if (tid == 0) {
        const float n = (float)(B_ * HW_);
        float mean = rs[0] / n;
        float var  = rq[0] / n - mean * mean;
        float r    = rsqrtf(var + 1e-5f);
        float g    = gamma[c] * r;
        scale[c] = g;
        shift[c] = beta[c] - mean * g;
    }
}

// ---------------------------------------------------------------------------
// K3: BN + SiLU, convert to f16
// ---------------------------------------------------------------------------
__global__ void bnsilu_kernel(const float* __restrict__ y1,
                              const float* __restrict__ scale,
                              const float* __restrict__ shift,
                              _Float16* __restrict__ t1) {
    int i = blockIdx.x * 256 + threadIdx.x;
    if (i >= B_ * MID_ * HW_) return;
    int c = (i >> 12) & (MID_ - 1);
    float v = y1[i] * scale[c] + shift[c];
    float sg = 1.f / (1.f + __expf(-v));
    t1[i] = (_Float16)(v * sg);
}

// ---------------------------------------------------------------------------
// K4: 3x3 conv WMMA GEMM  D[128p, 32768] = W[128p, 576'] * im2col
// Block = one full image row (64 pixels): 8 waves (m-tiles) x 4 n-subtiles.
// Raw 3-row slab staged in LDS channel-fastest; k' = tap*64+c makes both A and
// B fragment K-runs contiguous (-> b128 loads). No scattered im2col gathers.
// ---------------------------------------------------------------------------
__global__ __launch_bounds__(256) void conv3_kernel(const _Float16* __restrict__ t1,
                                                    const _Float16* __restrict__ w2,
                                                    float* __restrict__ y2) {
    __shared__ _Float16 sraw[3 * 66 * CSTR_];  // [(ry*66+col)][c], col 0 = x=-1 (zero)
    const int tid  = threadIdx.x;
    const int idx  = blockIdx.x;               // 0..511
    const int b    = idx >> 6;
    const int yrow = idx & 63;

    // stage transposed slab, coalesced along x per (c, ry)
    for (int e = tid; e < 64 * 3 * 66; e += 256) {
        int c   = e / 198;                     // 3*66
        int rem = e - c * 198;
        int ry  = rem / 66;
        int col = rem - ry * 66;
        int sy = yrow + ry - 1;
        int sx = col - 1;
        _Float16 v = (_Float16)0.f;
        if (sy >= 0 && sy < H_ && sx >= 0 && sx < W_)
            v = t1[(((size_t)b * MID_ + c) << 12) + (sy << 6) + sx];
        sraw[(ry * 66 + col) * CSTR_ + c] = v;
    }
    __syncthreads();

    const int lane = tid & 31;
    const int mt   = tid >> 5;
    const int row  = lane & 15;
    const int half = lane >> 4;
    const _Float16* wrow = w2 + (size_t)(mt * 16 + row) * K3_;
    __builtin_prefetch(wrow, 0, 1);            // global_prefetch_b8

    v8f acc[4];
#pragma unroll
    for (int nt = 0; nt < 4; ++nt) acc[nt] = (v8f){0.f,0.f,0.f,0.f,0.f,0.f,0.f,0.f};

    for (int ks = 0; ks < 18; ++ks) {
        const int k0 = ks * 32;
        v16h a;
#pragma unroll
        for (int v = 0; v < 8; ++v) {
            int ka = k0 + half * 16 + v * 2;
            h2 pa = *(const h2*)&wrow[ka];
            a[2 * v] = pa[0]; a[2 * v + 1] = pa[1];
        }
#pragma unroll
        for (int nt = 0; nt < 4; ++nt) {
            const int pix = nt * 16 + row;
            v16h bf;
#pragma unroll
            for (int v = 0; v < 8; ++v) {
                int k   = k0 + half * 16 + v * 2;
                int tap = k >> 6;              // constant within the 16-run
                int c   = k & 63;
                int ky  = tap / 3;
                int kx  = tap - ky * 3;
                h2 pb = *(const h2*)&sraw[(ky * 66 + pix + kx) * CSTR_ + c];
                bf[2 * v] = pb[0]; bf[2 * v + 1] = pb[1];
            }
            acc[nt] = __builtin_amdgcn_wmma_f32_16x16x32_f16(false, a, false, bf,
                                                             (short)0, acc[nt], false, false);
        }
    }
#pragma unroll
    for (int nt = 0; nt < 4; ++nt) {
#pragma unroll
        for (int r = 0; r < 8; ++r) {
            int m = mt * 16 + r + half * 8;
            if (m < OC_)
                y2[(((size_t)b * OC_ + m) << 12) + (yrow << 6) + nt * 16 + row] = acc[nt][r];
        }
    }
}

// ---------------------------------------------------------------------------
// K6: fused BN + pixel-shuffle + softmax(25) + CARAFE reassembly
// ---------------------------------------------------------------------------
__global__ __launch_bounds__(256) void carafe_out_kernel(const float* __restrict__ x,
                                                         const float* __restrict__ y2,
                                                         const float* __restrict__ scale2,
                                                         const float* __restrict__ shift2,
                                                         float* __restrict__ out) {
    __shared__ float slog[16 * 104];      // [pix][ch]
    __shared__ float swgt[16 * 104];      // [pix][tap*4+sub], f4-readable
    const int tid = threadIdx.x;
    const int idx = blockIdx.x;           // 0..2047
    const int b   = idx >> 8;
    const int rem = idx & 255;
    const int h   = rem >> 2;
    const int w0  = (rem & 3) << 4;

    for (int e = tid; e < 16 * OC_; e += 256) {
        int ch  = e >> 4;
        int pix = e & 15;
        float v = y2[(((size_t)b * OC_ + ch) << 12) + (h << 6) + w0 + pix];
        slog[pix * 104 + ch] = v * scale2[ch] + shift2[ch];
    }
    __syncthreads();

    if (tid < 64) {
        int pix = tid >> 2;
        int sub = tid & 3;
        float l[25];
        float mx = -1e30f;
#pragma unroll
        for (int k = 0; k < 25; ++k) {
            float v = slog[pix * 104 + k * 4 + sub];
            l[k] = v; mx = fmaxf(mx, v);
        }
        float s = 0.f;
#pragma unroll
        for (int k = 0; k < 25; ++k) { float e = __expf(l[k] - mx); l[k] = e; s += e; }
        float inv = 1.f / s;
#pragma unroll
        for (int k = 0; k < 25; ++k) swgt[pix * 104 + k * 4 + sub] = l[k] * inv;
    }
    __syncthreads();

    const int c  = tid >> 1;
    const int pb = (tid & 1) << 3;
    float acc[8][4];
#pragma unroll
    for (int i = 0; i < 8; ++i)
#pragma unroll
        for (int s = 0; s < 4; ++s) acc[i][s] = 0.f;

    const size_t xch = ((size_t)b * C_ + c) << 12;
    for (int dy = 0; dy < 5; ++dy) {
        int sy = h + dy - 2;
        float rowv[12];
#pragma unroll
        for (int j = 0; j < 12; ++j) {
            int sx = w0 + pb - 2 + j;
            rowv[j] = (sy >= 0 && sy < H_ && sx >= 0 && sx < W_)
                          ? x[xch + (sy << 6) + sx] : 0.f;
        }
#pragma unroll
        for (int i = 0; i < 8; ++i) {
            int pix = pb + i;
#pragma unroll
            for (int dx = 0; dx < 5; ++dx) {
                float xv = rowv[i + dx];
                int tap = dy * 5 + dx;
                f4 wv = *(const f4*)&swgt[pix * 104 + tap * 4];   // ds_load_b128
                acc[i][0] += wv[0] * xv;
                acc[i][1] += wv[1] * xv;
                acc[i][2] += wv[2] * xv;
                acc[i][3] += wv[3] * xv;
            }
        }
    }
    const size_t obase = ((size_t)b * C_ + c) << 14;
#pragma unroll
    for (int i = 0; i < 8; ++i) {
        int wcur = w0 + pb + i;
#pragma unroll
        for (int s = 0; s < 4; ++s) {
            int oy = 2 * h + (s >> 1);
            int ox = 2 * wcur + (s & 1);
            out[obase + (oy << 7) + ox] = acc[i][s];
        }
    }
}

// ---------------------------------------------------------------------------
extern "C" void kernel_launch(void* const* d_in, const int* in_sizes, int n_in,
                              void* d_out, int out_size, void* d_ws, size_t ws_size,
                              hipStream_t stream) {
    (void)in_sizes; (void)n_in; (void)out_size; (void)ws_size;
    const float* x      = (const float*)d_in[0];
    const float* comp_w = (const float*)d_in[1];
    const float* comp_g = (const float*)d_in[2];
    const float* comp_b = (const float*)d_in[3];
    const float* enc_w  = (const float*)d_in[4];
    const float* enc_g  = (const float*)d_in[5];
    const float* enc_b  = (const float*)d_in[6];
    float* out = (float*)d_out;

    char* ws = (char*)d_ws;
    float*    y1     = (float*)(ws);                  //  8,388,608 B
    _Float16* t1     = (_Float16*)(ws + 8388608);     //  4,194,304 B
    float*    y2     = (float*)(ws + 12582912);       // 13,107,200 B
    _Float16* w1h    = (_Float16*)(ws + 25690112);    //     16,384 B
    _Float16* w2h    = (_Float16*)(ws + 25706496);    //    147,456 B
    float*    scale1 = (float*)(ws + 25853952);
    float*    shift1 = scale1 + MID_;
    float*    scale2 = shift1 + MID_;
    float*    shift2 = scale2 + OC_;

    wconv_kernel<<<(OCP_ * K3_ + 255) / 256, 256, 0, stream>>>(comp_w, enc_w, w1h, w2h);
    conv1_kernel<<<NPIX_ / 128, 256, 0, stream>>>(x, w1h, y1);
    bnstats_kernel<<<MID_, 256, 0, stream>>>(y1, MID_, comp_g, comp_b, scale1, shift1);
    bnsilu_kernel<<<(B_ * MID_ * HW_ + 255) / 256, 256, 0, stream>>>(y1, scale1, shift1, t1);
    conv3_kernel<<<NPIX_ / 64, 256, 0, stream>>>(t1, w2h, y2);
    bnstats_kernel<<<OC_, 256, 0, stream>>>(y2, OC_, enc_g, enc_b, scale2, shift2);
    carafe_out_kernel<<<2048, 256, 0, stream>>>(x, y2, scale2, shift2, out);
}
